// GemmaAttention_20650202759999
// MI455X (gfx1250) — matrix operand order repaired and checked by reading, not compiled, not run on hardware
//
#include <hip/hip_runtime.h>
#include <hip/hip_bf16.h>

// ---------------------------------------------------------------------------
// CDNA5 (gfx1250) WMMA helpers: bf16 16x16x32, f32 accumulate, wave32
// ---------------------------------------------------------------------------
typedef __attribute__((ext_vector_type(16))) __bf16 bf16x16;
typedef __attribute__((ext_vector_type(8)))  float  f32x8;

union FragAB { bf16x16 v; unsigned int u[8]; };

__device__ inline unsigned short f2bf(float x) {
    unsigned int u = __float_as_uint(x);
    u += 0x7FFFu + ((u >> 16) & 1u);     // round-to-nearest-even
    return (unsigned short)(u >> 16);
}

// A-matrix fragment (16x32 bf16): lane L holds row m=L&15;
// lanes 0-15 cover K {0..7,16..23}, lanes 16-31 cover K {8..15,24..31}.
__device__ inline bf16x16 lds_frag_a(const unsigned short* base, int lda, int lane) {
    FragAB f;
    const int m  = lane & 15;
    const int kb = (lane >> 4) * 8;
    const unsigned short* row = base + m * lda;
#pragma unroll
    for (int j = 0; j < 8; ++j) {
        const int k = ((j >> 2) << 4) + kb + ((j & 3) << 1);
        f.u[j] = *(const unsigned int*)(row + k);
    }
    return f.v;
}

// B-matrix fragment (32x16 bf16), stored in LDS k-transposed: elem(k,n) at base[n*ldb + k].
__device__ inline bf16x16 lds_frag_b(const unsigned short* base, int ldb, int lane) {
    FragAB f;
    const int n  = lane & 15;
    const int kb = (lane >> 4) * 8;
    const unsigned short* row = base + n * ldb;
#pragma unroll
    for (int j = 0; j < 8; ++j) {
        f.u[j] = *(const unsigned int*)(row + ((j >> 2) << 4) + kb + ((j & 3) << 1));
    }
    return f.v;
}

__device__ inline f32x8 wmma_bf16(bf16x16 a, bf16x16 b, f32x8 c) {
    return __builtin_amdgcn_wmma_f32_16x16x32_bf16(false, a, false, b, (short)0, c,
                                                   false, false);
}

// ---------------------------------------------------------------------------
// CDNA5 async global->LDS copy (ASYNCcnt-tracked), 16 bytes per lane.
// VDST VGPR = LDS byte offset (LDS_BASE-relative); VADDR = 64-bit global addr.
// ---------------------------------------------------------------------------
__device__ inline void async_copy_b128(unsigned lds_byte_off, const void* g) {
    asm volatile("global_load_async_to_lds_b128 %0, %1, off"
                 :: "v"(lds_byte_off), "v"((unsigned long long)(uintptr_t)g)
                 : "memory");
}
__device__ inline void wait_async_le16() {
    asm volatile("s_wait_asynccnt 0x10" ::: "memory");
}
__device__ inline void wait_async_0() {
    asm volatile("s_wait_asynccnt 0x0" ::: "memory");
}

// ---------------------------------------------------------------------------
// Generic C[M,N] = A[M,K] @ B[K,N], fp32 in/out, bf16 WMMA math.
// Block tile 128x128x32; 8 waves (4x2); each wave 32x64 (8 x v8f accumulators).
// ---------------------------------------------------------------------------
__global__ __launch_bounds__(256) void gemm_bf16_wmma(
    const float* __restrict__ A, const float* __restrict__ B,
    float* __restrict__ C, int M, int N, int K) {
    __shared__ unsigned short Al[128 * 40];   // [m][k], pad 32->40
    __shared__ unsigned short Bl[128 * 40];   // transposed: [n][k]

    const int tid  = threadIdx.x;
    const int lane = tid & 31, wave = tid >> 5;
    const int wm = (wave >> 1) * 32;
    const int wn = (wave & 1) * 64;
    const int rowBase = blockIdx.y * 128;
    const int colBase = blockIdx.x * 128;

    f32x8 zero = {};
    f32x8 acc[2][4];
#pragma unroll
    for (int i = 0; i < 2; ++i)
#pragma unroll
        for (int j = 0; j < 4; ++j) acc[i][j] = zero;

    const int ar = tid >> 1, ac = (tid & 1) * 16;
    const int bn = tid >> 1, bk = (tid & 1) * 16;

    for (int k0 = 0; k0 < K; k0 += 32) {
        const float4* ag = (const float4*)(A + (size_t)(rowBase + ar) * K + k0 + ac);
#pragma unroll
        for (int j = 0; j < 4; ++j) {
            float4 t = ag[j];
            unsigned short* d = &Al[ar * 40 + ac + j * 4];
            d[0] = f2bf(t.x); d[1] = f2bf(t.y); d[2] = f2bf(t.z); d[3] = f2bf(t.w);
        }
        const float* bg = B + (size_t)(k0 + bk) * N + colBase + bn;
#pragma unroll
        for (int j = 0; j < 16; ++j)
            Bl[bn * 40 + bk + j] = f2bf(bg[(size_t)j * N]);
        __syncthreads();

        bf16x16 af[2], bfr[4];
#pragma unroll
        for (int mt = 0; mt < 2; ++mt)
            af[mt] = lds_frag_a(&Al[(wm + mt * 16) * 40], 40, lane);
#pragma unroll
        for (int nt = 0; nt < 4; ++nt)
            bfr[nt] = lds_frag_b(&Bl[(wn + nt * 16) * 40], 40, lane);
#pragma unroll
        for (int mt = 0; mt < 2; ++mt)
#pragma unroll
            for (int nt = 0; nt < 4; ++nt)
                acc[mt][nt] = wmma_bf16(af[mt], bfr[nt], acc[mt][nt]);
        __syncthreads();
    }

    const int n = lane & 15, mh = (lane >> 4) * 8;
#pragma unroll
    for (int mt = 0; mt < 2; ++mt)
#pragma unroll
        for (int nt = 0; nt < 4; ++nt)
#pragma unroll
            for (int r = 0; r < 8; ++r)
                C[(size_t)(rowBase + wm + mt * 16 + mh + r) * N +
                  colBase + wn + nt * 16 + n] = acc[mt][nt][r];
}

// ---------------------------------------------------------------------------
// RoPE: read fp32 X [rows][numHeads*256], write rotated bf16 Y (same layout).
// ---------------------------------------------------------------------------
__global__ void rope_bf16_kernel(const float* __restrict__ X,
                                 unsigned short* __restrict__ Y,
                                 int rows, int S, int numHeads) {
    long long idx = (long long)blockIdx.x * blockDim.x + threadIdx.x;
    long long total = (long long)rows * numHeads * 128;
    if (idx >= total) return;
    const int i = (int)(idx & 127);
    long long t = idx >> 7;
    const int h = (int)(t % numHeads);
    const long long row = t / numHeads;
    const int pos = (int)(row % S);
    const float inv = __expf(-((float)(2 * i) * (1.0f / 256.0f)) * 9.2103403719762f);
    const float f = (float)pos * inv;
    const float c = __cosf(f), s = __sinf(f);
    const long long base = row * ((long long)numHeads * 256) + h * 256;
    const float x1 = X[base + i], x2 = X[base + i + 128];
    Y[base + i]       = f2bf(x1 * c - x2 * s);
    Y[base + i + 128] = f2bf(x2 * c + x1 * s);
}

// ---------------------------------------------------------------------------
// V: fp32 [b*S+t][256] -> bf16 transposed [b][d=256][t=S] (so the flash kernel's
// LDS tile is a verbatim async copy, no in-kernel transpose needed).
// ---------------------------------------------------------------------------
__global__ void convert_v_t_kernel(const float* __restrict__ V,
                                   unsigned short* __restrict__ Vt,
                                   int S) {
    long long idx = (long long)blockIdx.x * blockDim.x + threadIdx.x;  // [row][d]
    const int d = (int)(idx & 255);
    const long long row = idx >> 8;            // b*S + t
    const long long b = row / S;
    const long long t = row - b * S;
    Vt[(b * 256 + d) * (long long)S + t] = f2bf(V[row * 256 + d]);
}

// ---------------------------------------------------------------------------
// Flash attention (causal, 1 KV head), bf16 WMMA math, f32 online softmax.
// bf16 operands async-copied from global to LDS with K/V double buffering.
// Grid: (S/128, numHeads, B). Block: 256 threads = 8 waves, 16 q-rows/wave.
// LDS (shorts): Q 32768 | K0 16384 | K1 16384 | V0 16384 | V1 16384 | P 9216
// ---------------------------------------------------------------------------
#define HD 256

__global__ __launch_bounds__(256) void flash_attn_wmma(
    const unsigned short* __restrict__ Qh, const unsigned short* __restrict__ Kh,
    const unsigned short* __restrict__ VtG, float* __restrict__ O,
    int S, int numHeads) {
    extern __shared__ char smem_raw[];
    unsigned short* sm = (unsigned short*)smem_raw;
    unsigned short* Ql  = sm;                 // [128][256]
    unsigned short* KlB[2] = { sm + 32768, sm + 49152 };   // [64][256]
    unsigned short* VtB[2] = { sm + 65536, sm + 81920 };   // [256][64]
    unsigned short* Pl  = sm + 98304;         // [8 waves][16][72]
    const unsigned ldsBase = (unsigned)(uintptr_t)sm;      // LDS aperture: addr[31:0]

    const int tid  = threadIdx.x;
    const int lane = tid & 31, wave = tid >> 5;
    const int qBase = blockIdx.x * 128;
    const int head  = blockIdx.y;
    const int b     = blockIdx.z;
    const long long rowOff = (long long)b * S;
    const int Hq = numHeads * HD;

    // ---- issue async copy of Q tile (128 x 256 bf16): 16 x b128 per thread
    {
        const int r = tid >> 1, hh = tid & 1;
        const unsigned short* g = Qh + (rowOff + qBase + r) * (long long)Hq +
                                  head * HD + hh * 128;
        const unsigned loff = ldsBase + (unsigned)((r * 256 + hh * 128) * 2);
#pragma unroll
        for (int j = 0; j < 16; ++j)
            async_copy_b128(loff + j * 16, g + j * 8);
    }

    // per-thread staging geometry for K (8 x b128) and V (8 x b128)
    const int kr = tid >> 2, kq = tid & 3;                       // K: row, 64-col quarter
    const unsigned kLoffRel = (unsigned)((kr * 256 + kq * 64) * 2);
    const unsigned vLoffRel = (unsigned)(tid * 64 * 2);          // V: one d-row per thread
    const unsigned short* vRow = VtG + ((long long)b * HD + tid) * (long long)S;

    // ---- issue async copy of KV tile 0
    {
        const unsigned short* kg = Kh + (rowOff + 0 + kr) * (long long)HD + kq * 64;
        const unsigned kl = ldsBase + (unsigned)((32768) * 2) + kLoffRel;
        const unsigned vl = ldsBase + (unsigned)((65536) * 2) + vLoffRel;
#pragma unroll
        for (int j = 0; j < 8; ++j) async_copy_b128(kl + j * 16, kg + j * 8);
#pragma unroll
        for (int j = 0; j < 8; ++j) async_copy_b128(vl + j * 16, vRow + 0 + j * 8);
    }

    f32x8 zero = {};
    f32x8 oacc[16];
#pragma unroll
    for (int i = 0; i < 16; ++i) oacc[i] = zero;
    float mrow[8], lrow[8];
#pragma unroll
    for (int r = 0; r < 8; ++r) { mrow[r] = -3.0e38f; lrow[r] = 0.0f; }

    const int nn = lane & 15, hsel = lane >> 4;
    unsigned short* PlW = Pl + wave * 16 * 72;

    const int numKv = (qBase >> 6) + 2;   // causal: tiles with any unmasked column
    for (int kt = 0; kt < numKv; ++kt) {
        const int kvBase = kt * 64;
        const int bi = kt & 1;

        // prefetch next KV tile into the alternate buffer, then wait for current
        if (kt + 1 < numKv) {
            const int nb = bi ^ 1;
            const int nkv = kvBase + 64;
            const unsigned short* kg = Kh + (rowOff + nkv + kr) * (long long)HD + kq * 64;
            const unsigned kl = ldsBase + (unsigned)((32768 + nb * 16384) * 2) + kLoffRel;
            const unsigned vl = ldsBase + (unsigned)((65536 + nb * 16384) * 2) + vLoffRel;
#pragma unroll
            for (int j = 0; j < 8; ++j) async_copy_b128(kl + j * 16, kg + j * 8);
#pragma unroll
            for (int j = 0; j < 8; ++j) async_copy_b128(vl + j * 16, vRow + nkv + j * 8);
            wait_async_le16();   // only the 16 next-tile copies may remain in flight
        } else {
            wait_async_0();
        }
        __syncthreads();

        const unsigned short* Kl = KlB[bi];
        const unsigned short* Vt = VtB[bi];

        // S = Q K^T : 16x64 per wave, head_dim in 8 chunks of 32
        f32x8 sc[4];
#pragma unroll
        for (int nt = 0; nt < 4; ++nt) sc[nt] = zero;
#pragma unroll
        for (int c = 0; c < 8; ++c) {
            bf16x16 a = lds_frag_a(&Ql[(wave * 16) * 256 + c * 32], 256, lane);
#pragma unroll
            for (int nt = 0; nt < 4; ++nt) {
                bf16x16 bb = lds_frag_b(&Kl[(nt * 16) * 256 + c * 32], 256, lane);
                sc[nt] = wmma_bf16(a, bb, sc[nt]);
            }
        }

        // online softmax: rows r+8*hsel, columns striped over 16 lanes
        float alpha[8];
#pragma unroll
        for (int r = 0; r < 8; ++r) {
            const int qRow = qBase + wave * 16 + r + 8 * hsel;
            float sv[4];
            float vmax = -3.0e38f;
#pragma unroll
            for (int nt = 0; nt < 4; ++nt) {
                const int col = kvBase + nt * 16 + nn;
                const float s_ = sc[nt][r] * 0.0625f;   // 1/sqrt(256)
                sv[nt] = (col <= qRow) ? s_ : -3.0e38f;
                vmax = fmaxf(vmax, sv[nt]);
            }
#pragma unroll
            for (int off = 1; off < 16; off <<= 1)
                vmax = fmaxf(vmax, __shfl_xor(vmax, off, 16));
            const float nm = fmaxf(mrow[r], vmax);
            alpha[r] = __expf(mrow[r] - nm);
            float rsum = 0.0f;
#pragma unroll
            for (int nt = 0; nt < 4; ++nt) {
                const float p = (sv[nt] > -1.0e37f) ? __expf(sv[nt] - nm) : 0.0f;
                rsum += p;
                PlW[(r + 8 * hsel) * 72 + nt * 16 + nn] = f2bf(p);
            }
#pragma unroll
            for (int off = 1; off < 16; off <<= 1)
                rsum += __shfl_xor(rsum, off, 16);
            lrow[r] = lrow[r] * alpha[r] + rsum;
            mrow[r] = nm;
        }
        // rescale running output
#pragma unroll
        for (int dt = 0; dt < 16; ++dt)
#pragma unroll
            for (int r = 0; r < 8; ++r) oacc[dt][r] *= alpha[r];

        // O += P @ V : P 16x64 (2 k-chunks), V^T tile gives B-frags per 16 d-cols
#pragma unroll
        for (int cc = 0; cc < 2; ++cc) {
            bf16x16 pa = lds_frag_a(PlW + cc * 32, 72, lane);
#pragma unroll
            for (int dt = 0; dt < 16; ++dt) {
                bf16x16 vb = lds_frag_b(&Vt[(dt * 16) * 64 + cc * 32], 64, lane);
                oacc[dt] = wmma_bf16(pa, vb, oacc[dt]);
            }
        }
        __syncthreads();
    }

    // epilogue: O /= l, write context [b*S+s][head*256 + d] as fp32
#pragma unroll
    for (int r = 0; r < 8; ++r) {
        const float inv = 1.0f / lrow[r];
        const long long row = rowOff + qBase + wave * 16 + r + 8 * hsel;
        float* op = O + row * (long long)Hq + head * HD;
#pragma unroll
        for (int dt = 0; dt < 16; ++dt)
            op[dt * 16 + nn] = oacc[dt][r] * inv;
    }
}

// ---------------------------------------------------------------------------
// Launch: QKV projections -> RoPE(bf16) + V convert -> flash attn -> O proj
// ---------------------------------------------------------------------------
extern "C" void kernel_launch(void* const* d_in, const int* in_sizes, int n_in,
                              void* d_out, int out_size, void* d_ws, size_t ws_size,
                              hipStream_t stream) {
    (void)in_sizes; (void)n_in; (void)out_size; (void)ws_size;
    const float* hs = (const float*)d_in[0];
    // d_in[1] = attention_mask — causal mask recomputed analytically in-kernel
    const float* Wq = (const float*)d_in[2];
    const float* Wk = (const float*)d_in[3];
    const float* Wv = (const float*)d_in[4];
    const float* Wo = (const float*)d_in[5];
    float* out = (float*)d_out;

    const int Bn = 2, S = 2048, H = 2048, NH = 8, D = 256;
    const int M = Bn * S;                               // 4096 rows

    float* qf  = (float*)d_ws;                          // [M][NH*D] fp32
    float* kf  = qf + (size_t)M * (NH * D);             // [M][D]    fp32
    float* vf  = kf + (size_t)M * D;                    // [M][D]    fp32
    float* ctx = vf + (size_t)M * D;                    // [M][NH*D] fp32
    unsigned short* qh = (unsigned short*)(ctx + (size_t)M * (NH * D)); // [M][NH*D] bf16
    unsigned short* kh = qh + (size_t)M * (NH * D);     // [M][D]   bf16 (rope'd)
    unsigned short* vt = kh + (size_t)M * D;            // [Bn][D][S] bf16 transposed

    dim3 blk(256);
    // projections (bf16 WMMA, f32 accumulate)
    gemm_bf16_wmma<<<dim3((NH * D) / 128, M / 128), blk, 0, stream>>>(hs, Wq, qf, M, NH * D, H);
    gemm_bf16_wmma<<<dim3(D / 128,        M / 128), blk, 0, stream>>>(hs, Wk, kf, M, D,      H);
    gemm_bf16_wmma<<<dim3(D / 128,        M / 128), blk, 0, stream>>>(hs, Wv, vf, M, D,      H);

    // RoPE -> bf16 operands; V -> bf16 transposed [b][d][t]
    const long long nrQ = (long long)M * NH * 128;
    rope_bf16_kernel<<<dim3((unsigned)((nrQ + 255) / 256)), blk, 0, stream>>>(qf, qh, M, S, NH);
    const long long nrK = (long long)M * 1 * 128;
    rope_bf16_kernel<<<dim3((unsigned)((nrK + 255) / 256)), blk, 0, stream>>>(kf, kh, M, S, 1);
    const long long nrV = (long long)M * D;
    convert_v_t_kernel<<<dim3((unsigned)((nrV + 255) / 256)), blk, 0, stream>>>(vf, vt, S);

    // flash attention: ~210 KB dynamic LDS (< 320 KB/WGP), async double-buffered KV
    const size_t shmem = (size_t)(32768 + 2 * 16384 + 2 * 16384 + 9216) * sizeof(unsigned short);
    flash_attn_wmma<<<dim3(S / 128, NH, Bn), blk, shmem, stream>>>(qh, kh, vt, ctx, S, NH);

    // output projection
    gemm_bf16_wmma<<<dim3(H / 128, M / 128), blk, 0, stream>>>(ctx, Wo, out, M, H, H);
}
